// DeformableTransformer_74655121539723
// MI455X (gfx1250) — matrix operand order repaired and checked by reading, not compiled, not run on hardware
//
#include <hip/hip_runtime.h>
#include <hip/hip_bf16.h>

typedef __attribute__((ext_vector_type(16))) _Float16 v16h;
typedef __attribute__((ext_vector_type(8)))  float    v8f;

#define B_Q   4
#define L_TOK 13294
#define M_ROW (B_Q * L_TOK)   // 53176
#define DMODEL 256
#define DFFN  1024

// ---------------------------------------------------------------------------
// Pack a row-major f32 weight [K][N] into the WMMA B-fragment layout:
// fragment element j of lane holds B[k0 + (lane>>4)*8 + (j&7) + (j>>3)*16][n0 + (lane&15)]
// stored so each lane reads one contiguous v16h (32B) per 32-K step.
// ---------------------------------------------------------------------------
__global__ void pack_w_f16(const float* __restrict__ W, _Float16* __restrict__ Wp,
                           int Kd, int Nd) {
    int i = blockIdx.x * blockDim.x + threadIdx.x;
    if (i >= Kd * Nd) return;
    int k = i / Nd, n = i % Nd;
    int kt  = k >> 5;          // 32-K tile
    int kl  = k & 31;
    int hlf = (kl >> 3) & 1;   // lane half
    int j   = (kl & 7) + ((kl >> 4) << 3);   // element within v16h
    int nt  = n >> 4;          // 16-N tile
    int ln  = hlf * 16 + (n & 15);
    Wp[(((size_t)kt * (Nd >> 4) + nt) * 32 + ln) * 16 + j] = (_Float16)W[i];
}

// q16 = f16(src + pos), mem16 = f16(src2 + memory_pos)
__global__ void addpos_f16(const float* __restrict__ a, const float* __restrict__ b,
                           const float* __restrict__ c, const float* __restrict__ d,
                           _Float16* __restrict__ q16, _Float16* __restrict__ m16,
                           long n) {
    long i = (long)blockIdx.x * blockDim.x + threadIdx.x;
    if (i < n) {
        q16[i] = (_Float16)(a[i] + b[i]);
        m16[i] = (_Float16)(c[i] + d[i]);
    }
}

// ---------------------------------------------------------------------------
// WMMA GEMM: C[M][N] = A[M][K] (f16 row-major) @ Wp (pre-packed f16) + bias.
// Block = 256 threads = 8 waves laid out 4(M) x 2(N): 64x32 block tile.
// Each wave: one 16x16 tile, K unrolled in 32-steps of v_wmma_f32_16x16x32_f16.
// A fragment (ISA 7.12.2, 16-bit A 16x32): lane holds row (lane&15); element
// pair (2t,2t+1) covers K = (lane>>4)*8 + 2t(+1) for t<4, +16 for t>=4 -> dword loads.
// ---------------------------------------------------------------------------
template<int K, bool RELU, bool F16OUT>
__global__ __launch_bounds__(256) void gemm_wmma_f16(
    const _Float16* __restrict__ A,
    const _Float16* __restrict__ Wp,
    const float*    __restrict__ bias,
    float*          __restrict__ C32,
    _Float16*       __restrict__ C16,
    int M, int N)
{
    const int lane  = threadIdx.x & 31;
    const int half  = lane >> 4;
    const int ln15  = lane & 15;
    const int wave  = threadIdx.x >> 5;
    const int waveM = wave >> 1;     // 0..3
    const int waveN = wave & 1;      // 0..1
    const int rowTile = blockIdx.x * 64 + waveM * 16;
    const int colTile = blockIdx.y * 32 + waveN * 16;
    const int nTiles  = N >> 4;
    const int ntile   = colTile >> 4;

    int mRow = rowTile + ln15;
    if (mRow >= M) mRow = M - 1;               // clamp; OOB rows never stored
    const _Float16* arow = A + (size_t)mRow * K;

    v8f acc = {0.f, 0.f, 0.f, 0.f, 0.f, 0.f, 0.f, 0.f};

#pragma unroll
    for (int kt = 0; kt < K / 32; ++kt) {
        union { v16h v; unsigned u[8]; } af;
        const _Float16* ak = arow + kt * 32 + half * 8;
#pragma unroll
        for (int t = 0; t < 8; ++t) {
            const int koff = (t < 4) ? (2 * t) : (16 + 2 * (t - 4));
            af.u[t] = *(const unsigned*)(ak + koff);
        }
        const v16h bf = *(const v16h*)(Wp + (((size_t)kt * nTiles + ntile) * 32 + lane) * 16);
        acc = __builtin_amdgcn_wmma_f32_16x16x32_f16(
                  false, af.v, false, bf, (short)0, acc, false, false);
    }

    const int   col = colTile + ln15;
    const float bv  = bias[col];
#pragma unroll
    for (int r = 0; r < 8; ++r) {
        const int row = rowTile + half * 8 + r;
        if (row < M) {
            float v = acc[r] + bv;
            if (RELU) v = v > 0.f ? v : 0.f;
            if (F16OUT) C16[(size_t)row * N + col] = (_Float16)v;
            else        C32[(size_t)row * N + col] = v;
        }
    }
}

// ---------------------------------------------------------------------------
// Multi-scale deformable sampling. One wave per (b, token, head); one lane per
// head-dim channel (HEAD_DIM == 32 == wave32). Scalar work (softmax of 16
// logits, reference point, bilinear weights) is computed redundantly per lane;
// the 128B f32 value gathers are lane-contiguous.
// ---------------------------------------------------------------------------
__global__ __launch_bounds__(256) void ms_deform_sample(
    const float* __restrict__ value,   // [B][L][256] f32
    const float* __restrict__ offs,    // [B*L][256]  f32
    const float* __restrict__ awl,     // [B*L][128]  f32 (logits)
    _Float16*    __restrict__ acc16)   // [B*L][256]  f16
{
    const int  lane = threadIdx.x & 31;
    const int  wave = threadIdx.x >> 5;
    const long item = (long)blockIdx.x * 8 + wave;     // ((b*L + q)*8 + h)
    const int  h  = (int)(item & 7);
    const long bq = item >> 3;
    const int  b  = (int)(bq / L_TOK);
    const int  q  = (int)(bq % L_TOK);

    // token's own level -> reference point
    int tH, tW, tS;
    if      (q < 10000) { tH = 100; tW = 100; tS = 0;     }
    else if (q < 12500) { tH = 50;  tW = 50;  tS = 10000; }
    else if (q < 13125) { tH = 25;  tW = 25;  tS = 12500; }
    else                { tH = 13;  tW = 13;  tS = 13125; }
    const int   idxl = q - tS;
    const int   rr = idxl / tW, rc = idxl % tW;
    const float rx = (rc + 0.5f) / (float)tW;
    const float ry = (rr + 0.5f) / (float)tH;

    // softmax over the 16 (level,point) logits of this head
    const float* ap = awl + bq * 128 + h * 16;
    float mx = ap[0];
#pragma unroll
    for (int i = 1; i < 16; ++i) mx = fmaxf(mx, ap[i]);
    float e[16], s = 0.f;
#pragma unroll
    for (int i = 0; i < 16; ++i) { e[i] = __expf(ap[i] - mx); s += e[i]; }
    const float inv = 1.f / s;

    const float* op = offs + bq * 256 + h * 32;
    const int LH[4] = {100, 50, 25, 13};
    const int LS[4] = {0, 10000, 12500, 13125};

    float a = 0.f;
#pragma unroll
    for (int l = 0; l < 4; ++l) {
        const int H_ = LH[l], W_ = LH[l], S_ = LS[l];
        const float* vl = value + ((size_t)b * L_TOK + S_) * 256 + h * 32 + lane;
#pragma unroll
        for (int p = 0; p < 4; ++p) {
            const float w  = e[l * 4 + p] * inv;
            const float lx = rx + op[(l * 4 + p) * 2 + 0] / (float)W_;
            const float ly = ry + op[(l * 4 + p) * 2 + 1] / (float)H_;
            const float x  = lx * (float)W_ - 0.5f;
            const float y  = ly * (float)H_ - 0.5f;
            const float x0 = floorf(x), y0 = floorf(y);
            const float wx1 = x - x0, wy1 = y - y0;
            const int ix0 = (int)x0, iy0 = (int)y0;
#pragma unroll
            for (int dy = 0; dy < 2; ++dy) {
                const int yi = iy0 + dy;
                if (yi < 0 || yi >= H_) continue;
                const float wy = dy ? wy1 : (1.f - wy1);
#pragma unroll
                for (int dx = 0; dx < 2; ++dx) {
                    const int xi = ix0 + dx;
                    if (xi < 0 || xi >= W_) continue;
                    const float wx = dx ? wx1 : (1.f - wx1);
                    a += w * wy * wx * vl[(size_t)(yi * W_ + xi) * 256];
                }
            }
        }
    }
    acc16[bq * 256 + h * 32 + lane] = (_Float16)a;
}

// out = LayerNorm(X + Y) * g + b ; optional f32 and f16 outputs
__global__ __launch_bounds__(256) void residual_ln(
    const float* __restrict__ X, const float* __restrict__ Y,
    const float* __restrict__ g, const float* __restrict__ bta,
    float* __restrict__ o32, _Float16* __restrict__ o16)
{
    const int row = blockIdx.x, tid = threadIdx.x;
    __shared__ float red[256];
    const float v = X[(size_t)row * 256 + tid] + Y[(size_t)row * 256 + tid];
    red[tid] = v; __syncthreads();
    for (int s = 128; s > 0; s >>= 1) { if (tid < s) red[tid] += red[tid + s]; __syncthreads(); }
    const float mu = red[0] * (1.f / 256.f);
    __syncthreads();
    const float c = v - mu;
    red[tid] = c * c; __syncthreads();
    for (int s = 128; s > 0; s >>= 1) { if (tid < s) red[tid] += red[tid + s]; __syncthreads(); }
    const float var = red[0] * (1.f / 256.f);
    const float o = c * rsqrtf(var + 1e-5f) * g[tid] + bta[tid];
    if (o32) o32[(size_t)row * 256 + tid] = o;
    if (o16) o16[(size_t)row * 256 + tid] = (_Float16)o;
}

// ---------------------------------------------------------------------------
extern "C" void kernel_launch(void* const* d_in, const int* in_sizes, int n_in,
                              void* d_out, int out_size, void* d_ws, size_t ws_size,
                              hipStream_t stream) {
    const float* src     = (const float*)d_in[0];
    const float* src2    = (const float*)d_in[1];
    const float* pos     = (const float*)d_in[2];
    const float* mpos    = (const float*)d_in[3];
    const float* value_w = (const float*)d_in[4];
    const float* value_b = (const float*)d_in[5];
    const float* off_w   = (const float*)d_in[6];
    const float* off_b   = (const float*)d_in[7];
    const float* aw_w    = (const float*)d_in[8];
    const float* aw_b    = (const float*)d_in[9];
    const float* out_w   = (const float*)d_in[10];
    const float* out_b   = (const float*)d_in[11];
    const float* ln_g    = (const float*)d_in[12];
    const float* ln_b    = (const float*)d_in[13];
    const float* lin1_w  = (const float*)d_in[14];
    const float* lin1_b  = (const float*)d_in[15];
    const float* lin2_w  = (const float*)d_in[16];
    const float* lin2_b  = (const float*)d_in[17];
    float* out = (float*)d_out;

    const int M = M_ROW;

    // workspace carve-up (256B aligned regions)
    size_t woff = 0;
    auto alloc = [&](size_t bytes) -> char* {
        char* p = (char*)d_ws + woff;
        woff += (bytes + 255) & ~(size_t)255;
        return p;
    };
    _Float16* wp_val = (_Float16*)alloc((size_t)256 * 256 * 2);
    _Float16* wp_off = (_Float16*)alloc((size_t)256 * 256 * 2);
    _Float16* wp_aw  = (_Float16*)alloc((size_t)256 * 128 * 2);
    _Float16* wp_out = (_Float16*)alloc((size_t)256 * 256 * 2);
    _Float16* wp_l1  = (_Float16*)alloc((size_t)256 * 1024 * 2);
    _Float16* wp_l2  = (_Float16*)alloc((size_t)1024 * 256 * 2);
    _Float16* q16    = (_Float16*)alloc((size_t)M * 256 * 2);
    _Float16* mem16  = (_Float16*)alloc((size_t)M * 256 * 2);
    float*    valbuf = (float*)   alloc((size_t)M * 256 * 4);
    float*    offbuf = (float*)   alloc((size_t)M * 256 * 4);
    float*    awbuf  = (float*)   alloc((size_t)M * 128 * 4);
    _Float16* h16    = (_Float16*)alloc((size_t)M * 1024 * 2);
    // dead-buffer reuse
    _Float16* acc16 = mem16;   // mem16 dead after value GEMM
    float*    attn  = offbuf;  // offs dead after sampling
    float*    xbuf  = valbuf;  // value dead after sampling
    _Float16* x16   = q16;     // q16 dead after off/aw GEMMs
    float*    ffn   = offbuf;  // attn dead after first LN

    // 1) pack all weights into WMMA B-fragment layout (f16)
    pack_w_f16<<<(256 * 256 + 255) / 256, 256, 0, stream>>>(value_w, wp_val, 256, 256);
    pack_w_f16<<<(256 * 256 + 255) / 256, 256, 0, stream>>>(off_w,   wp_off, 256, 256);
    pack_w_f16<<<(256 * 128 + 255) / 256, 256, 0, stream>>>(aw_w,    wp_aw,  256, 128);
    pack_w_f16<<<(256 * 256 + 255) / 256, 256, 0, stream>>>(out_w,   wp_out, 256, 256);
    pack_w_f16<<<(256 * 1024 + 255) / 256, 256, 0, stream>>>(lin1_w, wp_l1,  256, 1024);
    pack_w_f16<<<(1024 * 256 + 255) / 256, 256, 0, stream>>>(lin2_w, wp_l2, 1024, 256);

    // 2) q = src+pos, mem = src2+memory_pos (f16)
    {
        long n = (long)M * 256;
        addpos_f16<<<(unsigned)((n + 255) / 256), 256, 0, stream>>>(
            src, pos, src2, mpos, q16, mem16, n);
    }

    const dim3 blk(256);
    const dim3 g256((M + 63) / 64, DMODEL / 32);   // N = 256
    const dim3 g128((M + 63) / 64, 128 / 32);      // N = 128
    const dim3 gffn((M + 63) / 64, DFFN / 32);     // N = 1024

    // 3) projections (WMMA)
    gemm_wmma_f16<256, false, false><<<g256, blk, 0, stream>>>(
        mem16, wp_val, value_b, valbuf, nullptr, M, 256);
    gemm_wmma_f16<256, false, false><<<g256, blk, 0, stream>>>(
        q16, wp_off, off_b, offbuf, nullptr, M, 256);
    gemm_wmma_f16<256, false, false><<<g128, blk, 0, stream>>>(
        q16, wp_aw, aw_b, awbuf, nullptr, M, 128);

    // 4) deformable sampling -> acc16 (f16)
    ms_deform_sample<<<(unsigned)((size_t)M * 8 / 8), blk, 0, stream>>>(
        valbuf, offbuf, awbuf, acc16);

    // 5) output projection (WMMA)
    gemm_wmma_f16<256, false, false><<<g256, blk, 0, stream>>>(
        acc16, wp_out, out_b, attn, nullptr, M, 256);

    // 6) x = LN(src + attn)  (f32 for residual, f16 for FFN GEMM)
    residual_ln<<<M, blk, 0, stream>>>(src, attn, ln_g, ln_b, xbuf, x16);

    // 7) FFN (WMMA): h = relu(x @ lin1_w + b1) in f16; ffn = h @ lin2_w + b2
    gemm_wmma_f16<256, true, true><<<gffn, blk, 0, stream>>>(
        x16, wp_l1, lin1_b, nullptr, h16, M, 1024);
    gemm_wmma_f16<1024, false, false><<<g256, blk, 0, stream>>>(
        h16, wp_l2, lin2_b, ffn, nullptr, M, 256);

    // 8) out = LN(x + ffn)
    residual_ln<<<M, blk, 0, stream>>>(xbuf, ffn, ln_g, ln_b, out, nullptr);
}
